// ARMIN_72773925863759
// MI455X (gfx1250) — matrix operand order repaired and checked by reading, not compiled, not run on hardware
//
#include <hip/hip_runtime.h>
#include <hip/hip_bf16.h>

// ARMIN cell for MI455X (gfx1250, wave32, WMMA).
// Memory-bound on hmem (3x 256MB passes); GEMMs run on v_wmma_f32_16x16x32_bf16
// with bf16 hi/lo split (3 WMMAs) for ~fp32 accuracy. Streaming kernels use
// b128/b64 vector memory ops; h_index row staged to LDS via async-to-LDS copy.

typedef __attribute__((ext_vector_type(16))) __bf16 v16bf;
typedef __attribute__((ext_vector_type(8)))  __bf16 v8bf;
typedef __attribute__((ext_vector_type(8)))  float  v8f;

#define B_  256
#define X_  512
#define H_  1024
#define R_  512
#define M_  512

__device__ __forceinline__ float sigmf(float x){ return 1.0f/(1.0f+expf(-x)); }

// ---------------------------------------------------------------------------
// GEMM: C[m,n] = sum_k A[m,k]*W(k,n) + bias[n]
//   WT=false: W is (K x N) row-major ("NN")
//   WT=true : W is (N x K) row-major (C = A @ W^T, "NT")
//   ACT: 0 = none, 1 = sigmoid
// Block: 256 threads = 8 wave32; block tile 128x128, wave tile 32x64, K-step 32.
// ---------------------------------------------------------------------------
template<bool WT, int ACT>
__global__ __launch_bounds__(256) void gemm_bf16x3(
    const float* __restrict__ A, int lda,
    const float* __restrict__ W,
    const float* __restrict__ bias,
    float* __restrict__ C, int ldc,
    int N, int K)
{
  __shared__ __bf16 Ah[128][32] __attribute__((aligned(64)));
  __shared__ __bf16 Al[128][32] __attribute__((aligned(64)));
  __shared__ __bf16 Bh[128][32] __attribute__((aligned(64)));   // [n][k]
  __shared__ __bf16 Bl[128][32] __attribute__((aligned(64)));

  const int tid   = threadIdx.x;
  const int lane  = tid & 31;
  const int wave  = tid >> 5;
  const int lrow  = lane & 15;       // n (for B/C frags) or m (for A frags)
  const int lhalf = lane >> 4;       // selects K-half (A/B) or M-half (C)

  const int m0 = blockIdx.y * 128;
  const int n0 = blockIdx.x * 128;
  const int wm = (wave & 3) * 32;    // wave's m offset in block tile
  const int wn = (wave >> 2) * 64;   // wave's n offset in block tile

  v8f acc[2][4];
  #pragma unroll
  for (int i=0;i<2;++i)
    #pragma unroll
    for (int j=0;j<4;++j)
      #pragma unroll
      for (int e=0;e<8;++e) acc[i][j][e] = 0.0f;

  for (int k0 = 0; k0 < K; k0 += 32) {
    // ---- stage A tile: 128x32 floats = 1024 float4, 4 per thread ----
    #pragma unroll
    for (int i=0;i<4;++i){
      int idx = tid + i*256;              // float4 index
      int m  = idx >> 3;                  // 8 float4 per 32-wide row
      int k4 = (idx & 7) * 4;
      float4 v = *(const float4*)&A[(size_t)(m0+m)*lda + k0 + k4];
      float f[4] = {v.x, v.y, v.z, v.w};
      #pragma unroll
      for (int j=0;j<4;++j){
        __bf16 h = (__bf16)f[j];
        Ah[m][k4+j] = h;
        Al[m][k4+j] = (__bf16)(f[j] - (float)h);
      }
    }
    // ---- stage W tile into [n][k] layout ----
    #pragma unroll
    for (int i=0;i<4;++i){
      int idx = tid + i*256;              // float4 index
      if (WT){
        int n  = idx >> 3;
        int k4 = (idx & 7) * 4;
        float4 v = *(const float4*)&W[(size_t)(n0+n)*K + k0 + k4];
        float f[4] = {v.x, v.y, v.z, v.w};
        #pragma unroll
        for (int j=0;j<4;++j){
          __bf16 h = (__bf16)f[j];
          Bh[n][k4+j] = h;
          Bl[n][k4+j] = (__bf16)(f[j] - (float)h);
        }
      } else {
        int k  = idx >> 5;                // 32 float4 per 128-wide row
        int n4 = (idx & 31) * 4;
        float4 v = *(const float4*)&W[(size_t)(k0+k)*N + n0 + n4];
        float f[4] = {v.x, v.y, v.z, v.w};
        #pragma unroll
        for (int j=0;j<4;++j){            // transpose into [n][k]
          __bf16 h = (__bf16)f[j];
          Bh[n4+j][k] = h;
          Bl[n4+j][k] = (__bf16)(f[j] - (float)h);
        }
      }
    }
    __syncthreads();

    // ---- build WMMA fragments from LDS in contiguous K order (same for A and B) ----
    // A 16x32 bf16: lane holds row m=lrow; K = 16*lhalf + j (contiguous)
    // B 32x16 bf16 (stored [n][k]): lane holds col n=lrow; K = 16*lhalf + j (contiguous)
    const int kbA = lhalf * 16;
    const int kbB = lhalf * 16;
    v16bf afh[2], afl[2], bfh[4], bfl[4];
    #pragma unroll
    for (int mi=0;mi<2;++mi){
      int m = wm + mi*16 + lrow;
      v8bf c0 = *(const v8bf*)&Ah[m][kbA];
      v8bf c1 = *(const v8bf*)&Ah[m][kbA+8];
      v8bf d0 = *(const v8bf*)&Al[m][kbA];
      v8bf d1 = *(const v8bf*)&Al[m][kbA+8];
      #pragma unroll
      for (int j=0;j<8;++j){
        afh[mi][j] = c0[j]; afh[mi][j+8] = c1[j];
        afl[mi][j] = d0[j]; afl[mi][j+8] = d1[j];
      }
    }
    #pragma unroll
    for (int ni=0;ni<4;++ni){
      int n = wn + ni*16 + lrow;
      bfh[ni] = *(const v16bf*)&Bh[n][kbB];
      bfl[ni] = *(const v16bf*)&Bl[n][kbB];
    }

    // ---- bf16x3: AhBh + AhBl + AlBh ----
    #pragma unroll
    for (int mi=0;mi<2;++mi)
      #pragma unroll
      for (int ni=0;ni<4;++ni){
        acc[mi][ni] = __builtin_amdgcn_wmma_f32_16x16x32_bf16(
            false, afh[mi], false, bfh[ni], (short)0, acc[mi][ni], false, false);
        acc[mi][ni] = __builtin_amdgcn_wmma_f32_16x16x32_bf16(
            false, afh[mi], false, bfl[ni], (short)0, acc[mi][ni], false, false);
        acc[mi][ni] = __builtin_amdgcn_wmma_f32_16x16x32_bf16(
            false, afl[mi], false, bfh[ni], (short)0, acc[mi][ni], false, false);
      }
    __syncthreads();
  }

  // ---- epilogue: C/D layout: lane col n=lrow, VGPR r -> row r + 8*lhalf ----
  #pragma unroll
  for (int mi=0;mi<2;++mi)
    #pragma unroll
    for (int ni=0;ni<4;++ni){
      int n = n0 + wn + ni*16 + lrow;
      float bs = bias[n];
      #pragma unroll
      for (int r=0;r<8;++r){
        int m = m0 + wm + mi*16 + lhalf*8 + r;
        float v = acc[mi][ni][r] + bs;
        if (ACT == 1) v = sigmf(v);
        C[(size_t)m*ldc + n] = v;
      }
    }
}

// ---------------------------------------------------------------------------
// Elementwise / streaming kernels
// ---------------------------------------------------------------------------
__global__ __launch_bounds__(256) void build_concat_xc(
    const float* __restrict__ x, const float* __restrict__ c, float* __restrict__ cat)
{
  int i = blockIdx.x*256 + threadIdx.x;           // over B*1536
  int b = i / 1536, k = i % 1536;
  cat[(size_t)b*2048 + k] = (k < X_) ? x[(size_t)b*X_ + k]
                                     : c[(size_t)b*H_ + (k - X_)];
}

__global__ __launch_bounds__(256) void softmax512(float* __restrict__ d)
{
  __shared__ float red[256];
  const int t = threadIdx.x;
  float* row = d + (size_t)blockIdx.x * M_;
  float v0 = row[t], v1 = row[t+256];
  red[t] = fmaxf(v0, v1); __syncthreads();
  for (int s=128; s>0; s>>=1){ if (t<s) red[t] = fmaxf(red[t], red[t+s]); __syncthreads(); }
  float mx = red[0]; __syncthreads();
  float e0 = expf(v0-mx), e1 = expf(v1-mx);
  red[t] = e0 + e1; __syncthreads();
  for (int s=128; s>0; s>>=1){ if (t<s) red[t] += red[t+s]; __syncthreads(); }
  float inv = 1.0f / red[0];
  row[t] = e0*inv; row[t+256] = e1*inv;
}

// h_entry[b,r] = sum_m h_index[b,m]*hmem[b,m,r]  -> written into concat[:,1536:2048]
// One block per batch row; float2 (b64) streaming loads, dual accumulators.
// h_index row staged to LDS with CDNA5 async global->LDS copy (ASYNCcnt).
__global__ __launch_bounds__(256) void h_entry_kernel(
    const float* __restrict__ hmem, const float* __restrict__ hIndex,
    float* __restrict__ cat)
{
  __shared__ float hi[M_] __attribute__((aligned(16)));
  const int b = blockIdx.x;
  const int t = threadIdx.x;

  // 512 floats = 128 x b128 async transfers by lanes of the first 4 waves
  if (t < 128) {
    unsigned ldsoff = (unsigned)(uintptr_t)&hi[t*4];
    const float* gsrc = hIndex + (size_t)b*M_ + t*4;
    asm volatile("global_load_async_to_lds_b128 %0, %1, off"
                 :: "v"(ldsoff), "v"(gsrc) : "memory");
    asm volatile("s_wait_asynccnt 0" ::: "memory");
  }
  __syncthreads();

  const float2* base = (const float2*)hmem + (size_t)b*(M_*R_/2) + t;  // col pair 2t,2t+1
  float2 a0; a0.x = 0.0f; a0.y = 0.0f;
  float2 a1; a1.x = 0.0f; a1.y = 0.0f;
  for (int mo = 0; mo < M_; mo += 32) {
    if (mo + 64 <= M_)
      __builtin_prefetch(&base[(size_t)(mo+32)*(R_/2)], 0, 0);   // global_prefetch_b8
    #pragma unroll
    for (int mi = 0; mi < 32; mi += 2) {
      float2 v0 = base[(size_t)(mo+mi)*(R_/2)];
      float2 v1 = base[(size_t)(mo+mi+1)*(R_/2)];
      float w0 = hi[mo+mi], w1 = hi[mo+mi+1];
      a0.x = fmaf(w0, v0.x, a0.x); a0.y = fmaf(w0, v0.y, a0.y);
      a1.x = fmaf(w1, v1.x, a1.x); a1.y = fmaf(w1, v1.y, a1.y);
    }
  }
  float2 o; o.x = a0.x + a1.x; o.y = a0.y + a1.y;
  *(float2*)&cat[(size_t)b*2048 + 1536 + 2*t] = o;
}

__global__ __launch_bounds__(256) void apply_gate(
    const float* __restrict__ cat, const float* __restrict__ g, float* __restrict__ gated)
{
  int i = blockIdx.x*256 + threadIdx.x;           // over B*2048
  int b = i >> 11, k = i & 2047;
  float v = cat[i];
  if (k >= X_) v *= g[(size_t)b*1536 + (k - X_)];
  gated[i] = v;
}

__global__ __launch_bounds__(256) void lstm_gates(
    const float* __restrict__ pre, const float* __restrict__ c, float* __restrict__ out)
{
  int i = blockIdx.x*256 + threadIdx.x;           // over B*H
  int b = i >> 10, hh = i & 1023;
  const float* p = pre + (size_t)b*4608;
  float iv = p[hh], jv = p[1024+hh], fv = p[2048+hh], ov = p[3072+hh];
  float cc = c[i];
  float nc = tanhf(cc*sigmf(fv + 1.0f) + sigmf(iv)*tanhf(jv));
  float nh = nc * sigmf(ov);
  float* new_r = out;
  float* new_h = out + (size_t)B_*1536;
  float* new_c = new_h + (size_t)B_*H_;
  new_h[i] = nh;
  new_c[i] = nc;
  new_r[(size_t)b*1536 + hh] = nh;
}

__global__ __launch_bounds__(256) void r_out_kernel(
    const float* __restrict__ pre, const float* __restrict__ cat, float* __restrict__ out)
{
  int i = blockIdx.x*256 + threadIdx.x;           // over B*R
  int b = i >> 9, r = i & 511;
  float om = pre[(size_t)b*4608 + 4096 + r];
  float he = cat[(size_t)b*2048 + 1536 + r];
  out[(size_t)b*1536 + 1024 + r] = he * sigmf(om);   // new_r[:, H:]
}

__global__ __launch_bounds__(256) void build_concat2(
    const float* __restrict__ x, const float* __restrict__ new_c, float* __restrict__ c2)
{
  int i = blockIdx.x*256 + threadIdx.x;           // over B*1536
  int b = i / 1536, k = i % 1536;
  c2[i] = (k < X_) ? x[(size_t)b*X_ + k] : new_c[(size_t)b*H_ + (k - X_)];
}

// new_hmem = w_val[b,r]*h_index[b,m] + hmem*(1-h_index[b,m]); float4 streaming
__global__ __launch_bounds__(256) void hmem_update(
    const float4* __restrict__ hmem4, const float* __restrict__ hIndex,
    const float4* __restrict__ wval4, float4* __restrict__ out4)
{
  long i4 = (long)blockIdx.x*256 + threadIdx.x;   // over B*M*R/4
  int r4 = (int)(i4 & (R_/4 - 1));                // 128 float4 per r-row
  long bm = i4 >> 7;
  int m = (int)(bm & (M_-1));
  int b = (int)(bm >> 9);
  float wi = hIndex[(size_t)b*M_ + m];
  float om = 1.0f - wi;
  float4 hv = hmem4[i4];
  float4 wv = wval4[(size_t)b*(R_/4) + r4];
  float4 o;
  o.x = wv.x*wi + hv.x*om;
  o.y = wv.y*wi + hv.y*om;
  o.z = wv.z*wi + hv.z*om;
  o.w = wv.w*wi + hv.w*om;
  out4[i4] = o;
}

// ---------------------------------------------------------------------------
extern "C" void kernel_launch(void* const* d_in, const int* in_sizes, int n_in,
                              void* d_out, int out_size, void* d_ws, size_t ws_size,
                              hipStream_t stream) {
  (void)in_sizes; (void)n_in; (void)out_size; (void)ws_size;
  const float* x       = (const float*)d_in[0];
  // d_in[1] = h : unused by the reference computation
  const float* c       = (const float*)d_in[2];
  const float* hmem    = (const float*)d_in[3];
  const float* W_full  = (const float*)d_in[4];
  const float* bias    = (const float*)d_in[5];
  const float* W_full1 = (const float*)d_in[6];
  const float* bias1   = (const float*)d_in[7];
  const float* trans_w = (const float*)d_in[8];
  const float* trans_b = (const float*)d_in[9];
  const float* fc_w    = (const float*)d_in[10];
  const float* fc_b    = (const float*)d_in[11];
  float* out = (float*)d_out;

  float* ws      = (float*)d_ws;
  float* hIndex  = ws;  ws += (size_t)B_*M_;      // 256x512
  float* cat     = ws;  ws += (size_t)B_*2048;    // [x, c, h_entry]
  float* g       = ws;  ws += (size_t)B_*1536;
  float* gated   = ws;  ws += (size_t)B_*2048;
  float* pre     = ws;  ws += (size_t)B_*4608;
  float* cat2    = ws;  ws += (size_t)B_*1536;    // [x, new_c]
  float* wval    = ws;                            // 256x512

  // 1) concat[:, :1536] = [x, c]
  build_concat_xc<<<(B_*1536)/256, 256, 0, stream>>>(x, c, cat);
  // 2) read_head = [x,c] @ fc_w^T + fc_b   (NT, K=1536, N=512)
  gemm_bf16x3<true,0><<<dim3(M_/128, B_/128), 256, 0, stream>>>(
      cat, 2048, fc_w, fc_b, hIndex, M_, M_, 1536);
  // 3) h_index = softmax(read_head)
  softmax512<<<B_, 256, 0, stream>>>(hIndex);
  // 4) h_entry -> concat[:, 1536:2048]   (streams 256MB of hmem)
  h_entry_kernel<<<B_, 256, 0, stream>>>(hmem, hIndex, cat);
  // 5) g = sigmoid(concat @ W_full1 + bias1)  (NN, K=2048, N=1536, fused act)
  gemm_bf16x3<false,1><<<dim3(1536/128, B_/128), 256, 0, stream>>>(
      cat, 2048, W_full1, bias1, g, 1536, 1536, 2048);
  // 6) gated = concat * [ones, g]
  apply_gate<<<(B_*2048)/256, 256, 0, stream>>>(cat, g, gated);
  // 7) pre = gated @ W_full + bias  (NN, K=2048, N=4608 — the big GEMM)
  gemm_bf16x3<false,0><<<dim3(4608/128, B_/128), 256, 0, stream>>>(
      gated, 2048, W_full, bias, pre, 4608, 4608, 2048);
  // 8) LSTM elementwise -> new_h, new_c, new_r[:, :H] ; then r_out -> new_r[:, H:]
  lstm_gates<<<(B_*H_)/256, 256, 0, stream>>>(pre, c, out);
  r_out_kernel<<<(B_*R_)/256, 256, 0, stream>>>(pre, cat, out);
  // 9) concat2 = [x, new_c]
  const float* new_c_ptr = out + (size_t)B_*1536 + (size_t)B_*H_;
  build_concat2<<<(B_*1536)/256, 256, 0, stream>>>(x, new_c_ptr, cat2);
  // 10) w_val = concat2 @ trans_w^T + trans_b  (NT, K=1536, N=512)
  gemm_bf16x3<true,0><<<dim3(R_/128, B_/128), 256, 0, stream>>>(
      cat2, 1536, trans_w, trans_b, wval, R_, R_, 1536);
  // 11) new_hmem = lerp(hmem, w_val, h_index)  (streams 512MB, b128 ops)
  hmem_update<<<(B_*(size_t)M_*R_/4)/256, 256, 0, stream>>>(
      (const float4*)hmem, hIndex, (const float4*)wval,
      (float4*)(out + (size_t)B_*3584));
}
